// DiffAtten_36782099923673
// MI455X (gfx1250) — matrix operand order repaired and checked by reading, not compiled, not run on hardware
//
#include <hip/hip_runtime.h>
#include <hip/hip_bf16.h>

// ---------------- problem constants ----------------
#define B_  2
#define N_  1024
#define D_  64
#define H_  4
#define DK_ 16
#define L_  3072      // STEPS * N
#define DOUT_ 128

typedef __attribute__((ext_vector_type(16))) _Float16 v16h;
typedef __attribute__((ext_vector_type(8)))  float    v8f;

#define WMMA_F16(a, b, c) \
  __builtin_amdgcn_wmma_f32_16x16x32_f16(false, (a), false, (b), (short)0, (c), false, false)

// ------------- WMMA operand loaders (wave32, contiguous k = 16*(lane/16) + e, not the ISA table order) ---
// A-matrix 16x32 f16: lanes 0-15 row M=lane hold K 0..15 (elem e == K e);
// lanes 16-31 row M=lane-16 hold K 16..31.
__device__ __forceinline__ v16h load_a_f32(const float* __restrict__ A,
                                           int lda, int lane) {
  int row = lane & 15;
  int kb  = (lane < 16) ? 0 : 16;
  const float* p = A + (size_t)row * lda + kb;
  v16h a;
#pragma unroll
  for (int e = 0; e < 8; ++e) a[e]     = (_Float16)p[e];
#pragma unroll
  for (int e = 0; e < 8; ++e) a[8 + e] = (_Float16)p[8 + e];
  return a;
}

// B-matrix 32x16 f16: lane holds column N=lane&15; lanes 0-15 carry K 0..15
// (elem e == K e), lanes 16-31 carry K 16..31.
__device__ __forceinline__ v16h load_b_f32(const float* __restrict__ Bm,
                                           int ldb, int lane) {
  int col = lane & 15;
  int kb  = (lane < 16) ? 0 : 16;
  const float* p = Bm + (size_t)kb * ldb + col;
  v16h b;
#pragma unroll
  for (int e = 0; e < 16; ++e) b[e] = (_Float16)p[(size_t)e * ldb];
  return b;
}

// ---------------- generic f16-WMMA GEMM, fp32 in/out ------------------------
// 32x32 output per wave (2x2 WMMA register tile), K multiple of 32.
// C[m,n] = sum_k A[m,k]*B[k,n] (+bias[n]) (+add[m,n]) (relu?).
// blockIdx.z selects a batch slice by stride.
__global__ void gemm_wmma_f16(const float* __restrict__ A,
                              const float* __restrict__ Bm,
                              float* __restrict__ C,
                              int M, int K, int Nc,
                              long sA, long sB, long sC,
                              const float* __restrict__ bias,
                              const float* __restrict__ add, long sAdd,
                              int relu) {
  int lane = threadIdx.x & 31;
  int wave = threadIdx.x >> 5;
  int gn_cnt = Nc >> 5;
  int gm_cnt = M >> 5;
  int g = blockIdx.x * (blockDim.x >> 5) + wave;
  if (g >= gm_cnt * gn_cnt) return;         // wave-uniform: EXEC stays full
  int gm = g / gn_cnt, gn = g % gn_cnt;

  const float* Ab = A  + (size_t)blockIdx.z * sA + (size_t)gm * 32 * K;
  const float* Bb = Bm + (size_t)blockIdx.z * sB + gn * 32;

  // incremented prefetch pointers: next K-panel of A (32 rows) and B (32 rows)
  const float* pfA = Ab + (size_t)lane * K + 32;
  const float* pfB = Bb + (size_t)(32 + lane) * Nc;

  v8f acc00 = {}, acc01 = {}, acc10 = {}, acc11 = {};
  for (int k0 = 0; k0 < K; k0 += 32) {
    if (k0 + 32 < K) {                      // L2-resident: pull next panels near
      __builtin_prefetch(pfA, 0, 3);
      __builtin_prefetch(pfB, 0, 3);
    }
    pfA += 32;
    pfB += (size_t)32 * Nc;

    v16h a0 = load_a_f32(Ab + k0, K, lane);
    v16h a1 = load_a_f32(Ab + (size_t)16 * K + k0, K, lane);
    {   // phase 1: consume b0 fully before loading b1 (limit live fragments)
      v16h b0 = load_b_f32(Bb + (size_t)k0 * Nc, Nc, lane);
      acc00 = WMMA_F16(a0, b0, acc00);
      acc10 = WMMA_F16(a1, b0, acc10);
    }
    {   // phase 2
      v16h b1 = load_b_f32(Bb + (size_t)k0 * Nc + 16, Nc, lane);
      acc01 = WMMA_F16(a0, b1, acc01);
      acc11 = WMMA_F16(a1, b1, acc11);
    }
  }

  // C layout: VGPR r, lanes 0-15 -> M=r, lanes 16-31 -> M=8+r; N = lane&15.
  int crow0 = (lane < 16) ? 0 : 8;
  int ccol  = lane & 15;
  v8f* accs[2][2] = {{&acc00, &acc01}, {&acc10, &acc11}};
#pragma unroll
  for (int mt = 0; mt < 2; ++mt) {
#pragma unroll
    for (int nt = 0; nt < 2; ++nt) {
      int col = gn * 32 + nt * 16 + ccol;
      float bcol = bias ? bias[col] : 0.f;
      size_t rbase = (size_t)(gm * 32 + mt * 16);
      float* Cb = C + (size_t)blockIdx.z * sC + rbase * Nc + col;
      const float* Ad = add ? (add + (size_t)blockIdx.z * sAdd + rbase * Nc + col)
                            : nullptr;
#pragma unroll
      for (int r = 0; r < 8; ++r) {
        float v = (*accs[mt][nt])[r] + bcol;
        if (Ad)   v += Ad[(size_t)(crow0 + r) * Nc];
        if (relu) v = fmaxf(v, 0.f);
        Cb[(size_t)(crow0 + r) * Nc] = v;
      }
    }
  }
}

// ---------------- flash attention (one wave per 16-query tile) --------------
// Q,K,V: [B, L, 64] fp32, head h uses columns h*16..h*16+15. Online softmax,
// P tile transposed C-layout -> A-layout through per-wave LDS staging.
__global__ void flash_attn_wmma(const float* __restrict__ Qp,
                                const float* __restrict__ Kp,
                                const float* __restrict__ Vp,
                                float* __restrict__ O) {
  __shared__ _Float16 pbuf[4][16][40];   // row stride 80B (16B multiple -> b128)
  int lane = threadIdx.x & 31;
  int wave = threadIdx.x >> 5;
  int qt = blockIdx.x * 4 + wave;        // query tile 0..L/16-1
  int h  = blockIdx.y;
  int b  = blockIdx.z;
  size_t base = (size_t)b * L_ * D_ + h * DK_;

  // Q tile, A-layout, DK=16 zero-padded to K=32
  int arow = lane & 15;
  int d0   = 0;
  const float* qptr = Qp + base + (size_t)(qt * 16 + arow) * D_ + d0;
  v16h qa;
#pragma unroll
  for (int e = 0; e < 8; ++e)  qa[e] = (lane < 16) ? (_Float16)qptr[e] : (_Float16)0.f;
#pragma unroll
  for (int e = 8; e < 16; ++e) qa[e] = (lane < 16) ? (_Float16)qptr[e] : (_Float16)0.f;

  v8f acc = {}, lsum = {}, mrow;
#pragma unroll
  for (int r = 0; r < 8; ++r) mrow[r] = -1e30f;

  // incremented prefetch pointers: next 32-key block of K and V
  const float* pfK = Kp + base + (size_t)(32 + lane) * D_;
  const float* pfV = Vp + base + (size_t)(32 + lane) * D_;

  for (int kb = 0; kb < L_; kb += 32) {
    if (kb + 32 < L_) {
      __builtin_prefetch(pfK, 0, 3);
      __builtin_prefetch(pfV, 0, 3);
    }
    pfK += 32 * D_;
    pfV += 32 * D_;

    // ---- scores for two 16-key tiles: S = Q @ K^T, B[k=d][n=key] = K[key,d]
    v8f s[2];
#pragma unroll
    for (int t = 0; t < 2; ++t) {
      v16h kv;
      int n = lane & 15;
      if (lane < 16) {                       // K(dim) 0..15 valid
        const float* kp = Kp + base + (size_t)(kb + t * 16 + n) * D_;
#pragma unroll
        for (int e = 0; e < 16; ++e) kv[e] = (_Float16)kp[e];
      } else {                               // K(dim) 16..31 -> zero pad
#pragma unroll
        for (int e = 0; e < 16; ++e) kv[e] = (_Float16)0.f;
      }
      v8f z = {};
      s[t] = WMMA_F16(qa, kv, z);
#pragma unroll
      for (int r = 0; r < 8; ++r) s[t][r] *= 0.25f;   // 1/sqrt(DK)
    }

    // ---- online softmax over this 32-key block (row = VGPR r within half)
    v8f mnew = mrow;
#pragma unroll
    for (int r = 0; r < 8; ++r) {
      float mx = fmaxf(s[0][r], s[1][r]);
      mx = fmaxf(mx, __shfl_xor(mx, 1, 32));
      mx = fmaxf(mx, __shfl_xor(mx, 2, 32));
      mx = fmaxf(mx, __shfl_xor(mx, 4, 32));
      mx = fmaxf(mx, __shfl_xor(mx, 8, 32));
      mnew[r] = fmaxf(mnew[r], mx);
    }
#pragma unroll
    for (int r = 0; r < 8; ++r) {
      float alpha = __expf(mrow[r] - mnew[r]);
      acc[r]  *= alpha;
      lsum[r] *= alpha;
      mrow[r]  = mnew[r];
    }

    int prow0 = (lane < 16) ? 0 : 8;
    int pcol  = lane & 15;
#pragma unroll
    for (int r = 0; r < 8; ++r) {
      float p0 = __expf(s[0][r] - mrow[r]);
      float p1 = __expf(s[1][r] - mrow[r]);
      float rs = p0 + p1;
      rs += __shfl_xor(rs, 1, 32);
      rs += __shfl_xor(rs, 2, 32);
      rs += __shfl_xor(rs, 4, 32);
      rs += __shfl_xor(rs, 8, 32);
      lsum[r] += rs;
      pbuf[wave][prow0 + r][pcol]      = (_Float16)p0;  // keys kb..kb+15
      pbuf[wave][prow0 + r][16 + pcol] = (_Float16)p1;  // keys kb+16..kb+31
    }
    asm volatile("s_wait_dscnt 0" ::: "memory");  // same-wave LDS RAW

    // ---- P (16x32) in A-layout from LDS (two aligned 16B reads per lane)
    v16h pa;
    int pk = (lane < 16) ? 0 : 16;
#pragma unroll
    for (int e = 0; e < 8; ++e) pa[e]     = pbuf[wave][arow][pk + e];
#pragma unroll
    for (int e = 0; e < 8; ++e) pa[8 + e] = pbuf[wave][arow][pk + 8 + e];

    // ---- V (32x16) in B-layout: B[k=key][n=vdim] = V[key, vdim]
    v16h vb;
    {
      int n  = lane & 15;
      int kk = (lane < 16) ? 0 : 16;
      const float* vp = Vp + base + (size_t)(kb + kk) * D_ + n;
#pragma unroll
      for (int e = 0; e < 16; ++e) vb[e] = (_Float16)vp[(size_t)e * D_];
    }
    acc = WMMA_F16(pa, vb, acc);
  }

  int crow0 = (lane < 16) ? 0 : 8;
  int ccol  = lane & 15;
  float* op = O + base + (size_t)(qt * 16) * D_ + ccol;
#pragma unroll
  for (int r = 0; r < 8; ++r)
    op[(size_t)(crow0 + r) * D_] = acc[r] / lsum[r];
}

// ---------------- LayerNorm over 64-wide rows (one wave per row) ------------
__global__ void ln64(const float* __restrict__ in, float* __restrict__ out,
                     const float* __restrict__ w, const float* __restrict__ b,
                     int rows, float eps) {
  int row = blockIdx.x * (blockDim.x >> 5) + (threadIdx.x >> 5);
  if (row >= rows) return;
  int lane = threadIdx.x & 31;
  const float* p = in + (size_t)row * 64;
  float x0 = p[lane], x1 = p[lane + 32];
  float s = x0 + x1;
#pragma unroll
  for (int m = 16; m >= 1; m >>= 1) s += __shfl_xor(s, m, 32);
  float mu = s * (1.f / 64.f);
  float d0 = x0 - mu, d1 = x1 - mu;
  float v = d0 * d0 + d1 * d1;
#pragma unroll
  for (int m = 16; m >= 1; m >>= 1) v += __shfl_xor(v, m, 32);
  float inv = rsqrtf(v * (1.f / 64.f) + eps);
  float y0 = d0 * inv, y1 = d1 * inv;
  if (w) { y0 = y0 * w[lane] + b[lane]; y1 = y1 * w[lane + 32] + b[lane + 32]; }
  float* q = out + (size_t)row * 64;
  q[lane] = y0; q[lane + 32] = y1;
}

// ---------------- seed copy: q_x/v_x -> row-block 0 of qcat/vcat ------------
__global__ void seed_copy(const float* __restrict__ qx,
                          const float* __restrict__ vx,
                          float* __restrict__ qcat, float* __restrict__ vcat) {
  int idx = blockIdx.x * blockDim.x + threadIdx.x;
  if (idx >= B_ * N_ * D_) return;
  int b = idx / (N_ * D_);
  int r = idx % (N_ * D_);
  qcat[(size_t)b * L_ * D_ + r] = qx[idx];
  vcat[(size_t)b * L_ * D_ + r] = vx[idx];
}

// ---------------- pool: x[b,i,j] = mean_s o2_flat[b, 192*i + 3*j + s] -------
__global__ void pool3(const float* __restrict__ o2, float* __restrict__ x) {
  int idx = blockIdx.x * blockDim.x + threadIdx.x;
  if (idx >= B_ * N_ * D_) return;
  int b = idx / (N_ * D_);
  int rem = idx % (N_ * D_);
  int i = rem >> 6, j = rem & 63;
  const float* p = o2 + (size_t)b * L_ * D_ + 192 * i + 3 * j;
  x[idx] = (p[0] + p[1] + p[2]) * (1.f / 3.f);
}

// ---------------------------------------------------------------------------
extern "C" void kernel_launch(void* const* d_in, const int* in_sizes, int n_in,
                              void* d_out, int out_size, void* d_ws, size_t ws_size,
                              hipStream_t stream) {
  const float* q_x      = (const float*)d_in[0];
  const float* q_adj    = (const float*)d_in[1];
  const float* v_x      = (const float*)d_in[2];
  const float* v_adj    = (const float*)d_in[3];
  const float* w_qs     = (const float*)d_in[4];
  const float* w_ks     = (const float*)d_in[5];
  const float* w_vs     = (const float*)d_in[6];
  const float* w_fc     = (const float*)d_in[7];
  const float* mha_ln_w = (const float*)d_in[8];
  const float* mha_ln_b = (const float*)d_in[9];
  const float* conv_w   = (const float*)d_in[10];
  const float* conv_b   = (const float*)d_in[11];
  const float* lin_w    = (const float*)d_in[12];
  const float* lin_b    = (const float*)d_in[13];
  const float* norm_w   = (const float*)d_in[14];
  const float* norm_b   = (const float*)d_in[15];
  float* out = (float*)d_out;

  const size_t SLD = (size_t)B_ * L_ * D_;   // 393216
  float* w    = (float*)d_ws;
  float* qcat = w;                  // [B, L, D]
  float* vcat = qcat + SLD;
  float* qc   = vcat + SLD;
  float* vc   = qc + SLD;
  float* Qp   = vc + SLD;
  float* Kp   = Qp + SLD;
  float* Vp   = Kp + SLD;
  float* xb   = Vp + SLD;           // [B, N, D]
  float* x1   = xb + (size_t)B_ * N_ * D_;  // [B, N, DOUT]
  float* Obuf = qcat;               // reuse after LN
  float* o2   = vcat;               // reuse after LN

  dim3 blk(128);

  // 1) seed diffusion step 0
  seed_copy<<<dim3(512), dim3(256), 0, stream>>>(q_x, v_x, qcat, vcat);

  // 2) diffusion steps 1 and 2:  rows[N..2N) = A @ rows[0..N), etc.
  //    32x32 wave tiles: (1024/32)*(64/32) = 64 waves -> 16 blocks, z = B
  const long sAdj = (long)N_ * N_, sCat = (long)L_ * D_;
  for (int step = 1; step < 3; ++step) {
    const float* qsrc = qcat + (size_t)(step - 1) * N_ * D_;
    const float* vsrc = vcat + (size_t)(step - 1) * N_ * D_;
    gemm_wmma_f16<<<dim3(16, 1, B_), blk, 0, stream>>>(
        q_adj, qsrc, qcat + (size_t)step * N_ * D_,
        N_, N_, D_, sAdj, sCat, sCat, nullptr, nullptr, 0, 0);
    gemm_wmma_f16<<<dim3(16, 1, B_), blk, 0, stream>>>(
        v_adj, vsrc, vcat + (size_t)step * N_ * D_,
        N_, N_, D_, sAdj, sCat, sCat, nullptr, nullptr, 0, 0);
  }

  // 3) LN (eps 1e-5, no affine) on concatenated diffusion stacks
  ln64<<<dim3(1536), blk, 0, stream>>>(qcat, qc, nullptr, nullptr, B_ * L_, 1e-5f);
  ln64<<<dim3(1536), blk, 0, stream>>>(vcat, vc, nullptr, nullptr, B_ * L_, 1e-5f);

  // 4) QKV projections: [B*L,64] @ [64,64] -> (6144/32)*(64/32)=384 waves
  gemm_wmma_f16<<<dim3(96), blk, 0, stream>>>(qc, w_qs, Qp, B_ * L_, D_, D_,
                                              0, 0, 0, nullptr, nullptr, 0, 0);
  gemm_wmma_f16<<<dim3(96), blk, 0, stream>>>(qc, w_ks, Kp, B_ * L_, D_, D_,
                                              0, 0, 0, nullptr, nullptr, 0, 0);
  gemm_wmma_f16<<<dim3(96), blk, 0, stream>>>(vc, w_vs, Vp, B_ * L_, D_, D_,
                                              0, 0, 0, nullptr, nullptr, 0, 0);

  // 5) flash attention: grid (L/16/4 q-tile groups, H, B)
  flash_attn_wmma<<<dim3(L_ / 16 / 4, H_, B_), blk, 0, stream>>>(Qp, Kp, Vp, Obuf);

  // 6) fc + residual(qc), then MHA LN (eps 1e-6, affine) in place
  gemm_wmma_f16<<<dim3(96), blk, 0, stream>>>(Obuf, w_fc, o2, B_ * L_, D_, D_,
                                              0, 0, 0, nullptr, qc, 0, 0);
  ln64<<<dim3(1536), blk, 0, stream>>>(o2, o2, mha_ln_w, mha_ln_b, B_ * L_, 1e-6f);

  // 7) pool across diffusion steps (reshape-faithful)
  pool3<<<dim3(512), dim3(256), 0, stream>>>(o2, xb);

  // 8) conv: relu(x @ [64,128] + b): (2048/32)*(128/32)=256 waves
  gemm_wmma_f16<<<dim3(64), blk, 0, stream>>>(xb, conv_w, x1, B_ * N_, D_, DOUT_,
                                              0, 0, 0, conv_b, nullptr, 0, 1);

  // 9) lin: q_x + x1 @ [128,64] + b -> d_out: (2048/32)*(64/32)=128 waves
  gemm_wmma_f16<<<dim3(32), blk, 0, stream>>>(x1, lin_w, out, B_ * N_, DOUT_, D_,
                                              0, 0, 0, lin_b, q_x, 0, 0);
  ln64<<<dim3(512), blk, 0, stream>>>(out, out, norm_w, norm_b, B_ * N_, 1e-5f);
}
